// TemporalFusion_65094524338411
// MI455X (gfx1250) — compile-verified, matrix operand order checked by reading, not run on hardware
//
#include <hip/hip_runtime.h>
#include <math.h>
#include <cstdint>

#define USE_ASYNC_LDS 1   // gfx1250 global_load_async_to_lds_b128 + s_wait_asynccnt

// ---------------- model constants ----------------
#define CDIM 256
#define TT   5
#define KW   3
#define K2   9
#define LK   (TT*K2)        // 45 kv tokens
#define HD   32             // head dim
#define BB   4
#define HH   32
#define WW   32
#define NPIX (BB*HH*WW)     // 4096 query tokens
#define MFEAT (BB*TT*HH*WW) // 20480 projected feature rows

typedef __attribute__((ext_vector_type(16))) _Float16 v16h;
typedef __attribute__((ext_vector_type(8)))  _Float16 v8h;
typedef __attribute__((ext_vector_type(8)))  float    v8f;

// =====================================================================
// prep: feats (B,T,C,H,W) f32 -> featsT (B*T*H*W, C) f16 (GEMM-ready),
//       x0 f32 + xh0 f16 = centre-frame query tokens
// =====================================================================
__global__ __launch_bounds__(256)
void prep_kernel(const float* __restrict__ feats,
                 _Float16* __restrict__ featsT,
                 float* __restrict__ x0, _Float16* __restrict__ xh0)
{
    long i = (long)blockIdx.x * blockDim.x + threadIdx.x;   // src-linear: coalesced reads
    if (i >= (long)BB*TT*CDIM*HH*WW) return;
    int w  = (int)(i & (WW-1));
    int h  = (int)((i >> 5) & (HH-1));
    long r = i >> 10;                 // (bt, c)
    int c  = (int)(r % CDIM);
    long bt = r / CDIM;
    float v = feats[i];
    long m = bt * (HH*WW) + h*WW + w;
    featsT[m*CDIM + c] = (_Float16)v;
    if ((int)(bt % TT) == TT/2) {
        long p = ((bt/TT)*(HH*WW) + h*WW + w)*CDIM + c;
        x0[p]  = v;
        xh0[p] = (_Float16)v;
    }
}

// generic f32 -> f16 weight conversion
__global__ __launch_bounds__(256)
void cvt_f16_kernel(const float* __restrict__ src, _Float16* __restrict__ dst, int n)
{
    int i = blockIdx.x * 256 + threadIdx.x;
    if (i < n) dst[i] = (_Float16)src[i];
}

// posmat_h[tk, c] = tp[t, c] + sp[k, c]   (tk = t*9 + k), f16
__global__ __launch_bounds__(256)
void posmat_kernel(const float* __restrict__ tp, const float* __restrict__ sp,
                   _Float16* __restrict__ posmat)
{
    int i = blockIdx.x * blockDim.x + threadIdx.x;
    if (i >= LK*CDIM) return;
    int c = i & (CDIM-1);
    int tk = i >> 8;
    posmat[i] = (_Float16)(tp[(tk/K2)*CDIM + c] + sp[(tk%K2)*CDIM + c]);
}

// =====================================================================
// WMMA GEMM: C[M,N] = A[M,K](f16) * W[N,K](f16)^T + bias(f32)
// 128x64 tile / block, 8 waves (4 x M, 2 x N); each wave owns a 32x32
// macro-tile = 4 x v_wmma_f32_16x16x32_f16 per 32-wide K step
// (8 ds_load_b128 -> 4 WMMAs). Double-buffered LDS staging via gfx1250
// async global->LDS b128 copies; single barrier per K-step:
// s_wait_asynccnt 0 + barrier covers both the RAW on buf^1 and (via the
// barrier's DScnt drain) the WAR on buf.
// Fragment gather per CDNA5 ISA 7.12.2 16-bit layouts:
//   lanes 0-15: halves 0..7 -> K0..7,  8..15 -> K16..23
//   lanes16-31: halves 0..7 -> K8..15, 8..15 -> K24..31
// =====================================================================
__device__ __forceinline__ float gelu_exact(float v) {
    return 0.5f * v * (1.0f + erff(v * 0.70710678118654752f));
}

template<bool OUT_F16, int ACT, bool GUARD>
__global__ __launch_bounds__(256)
void gemm_wmma(const _Float16* __restrict__ A, const _Float16* __restrict__ Wh,
               const float* __restrict__ bias, void* __restrict__ Cptr,
               int M, int N, int K)
{
    constexpr int LDT = 48;                 // 96B row stride: 16B-aligned rows
    __shared__ _Float16 As[2][128*LDT];     // 2 x 12 KiB
    __shared__ _Float16 Bs[2][ 64*LDT];     // 2 x  6 KiB

    const int tid  = threadIdx.x;
    const int lane = tid & 31;
    const int wid  = tid >> 5;
    const int wm   = wid & 3;               // wave row block (32 rows)
    const int wn   = wid >> 2;              // wave col block (32 cols)
    const int lrow = lane & 15;
    const int lhi  = lane >> 4;
    const int mtile = blockIdx.y * 128;
    const int ntile = blockIdx.x * 64;

    // staging map: thread -> (row, 8-half column group); A covers rows r and r+64
    const int r  = tid >> 2;                // 0..63
    const int cg = (tid & 3) << 3;          // 0,8,16,24

    const _Float16* gA0 = A  + (size_t)(mtile + r) * K + cg;
    const _Float16* gA1 = gA0 + (size_t)64 * K;
    const _Float16* gB  = Wh + (size_t)(ntile + r) * K + cg;

    auto stage = [&](int bufsel, int kb) {
        if (GUARD) {                        // tiny-M path (M=45): sync, row-guarded
            v8h av0 = {}, av1 = {};
            if (mtile + r      < M) av0 = *(const v8h*)(gA0 + kb);
            if (mtile + r + 64 < M) av1 = *(const v8h*)(gA1 + kb);
            *(v8h*)&As[bufsel][ r      *LDT + cg] = av0;
            *(v8h*)&As[bufsel][(r + 64)*LDT + cg] = av1;
            *(v8h*)&Bs[bufsel][ r      *LDT + cg] = *(const v8h*)(gB + kb);
        } else {
#if USE_ASYNC_LDS
            uint32_t la0 = (uint32_t)(uintptr_t)&As[bufsel][ r      *LDT + cg];
            uint32_t la1 = (uint32_t)(uintptr_t)&As[bufsel][(r + 64)*LDT + cg];
            uint32_t lb  = (uint32_t)(uintptr_t)&Bs[bufsel][ r      *LDT + cg];
            const void* pa0 = (const void*)(gA0 + kb);
            const void* pa1 = (const void*)(gA1 + kb);
            const void* pb  = (const void*)(gB  + kb);
            asm volatile("global_load_async_to_lds_b128 %0, %3, off\n\t"
                         "global_load_async_to_lds_b128 %1, %4, off\n\t"
                         "global_load_async_to_lds_b128 %2, %5, off"
                         :: "v"(la0), "v"(la1), "v"(lb),
                            "v"(pa0), "v"(pa1), "v"(pb) : "memory");
#else
            *(v8h*)&As[bufsel][ r      *LDT + cg] = *(const v8h*)(gA0 + kb);
            *(v8h*)&As[bufsel][(r + 64)*LDT + cg] = *(const v8h*)(gA1 + kb);
            *(v8h*)&Bs[bufsel][ r      *LDT + cg] = *(const v8h*)(gB  + kb);
#endif
        }
    };

    v8f c00 = {}, c01 = {}, c10 = {}, c11 = {};

    stage(0, 0);
    int buf = 0;
    const int arow0 = (wm * 32 +      lrow) * LDT;
    const int arow1 = (wm * 32 + 16 + lrow) * LDT;
    const int brow0 = (wn * 32 +      lrow) * LDT;
    const int brow1 = (wn * 32 + 16 + lrow) * LDT;

    for (int kb = 0; kb < K; kb += 32) {
#if USE_ASYNC_LDS
        if (!GUARD) asm volatile("s_wait_asynccnt 0x0" ::: "memory");
#endif
        __syncthreads();                          // publish 'buf'; frees 'buf^1'
        if (kb + 32 < K) stage(buf ^ 1, kb + 32); // overlap next-tile copy with WMMA

        v16h a0, a1, b0, b1;
        #pragma unroll
        for (int i = 0; i < 16; ++i) {
            int k = (i < 8 ? i : i + 8) + (lhi << 3);
            a0[i] = As[buf][arow0 + k];
            a1[i] = As[buf][arow1 + k];
            b0[i] = Bs[buf][brow0 + k];
            b1[i] = Bs[buf][brow1 + k];
        }
        c00 = __builtin_amdgcn_wmma_f32_16x16x32_f16(false, a0, false, b0, (short)0, c00, false, false);
        c01 = __builtin_amdgcn_wmma_f32_16x16x32_f16(false, a0, false, b1, (short)0, c01, false, false);
        c10 = __builtin_amdgcn_wmma_f32_16x16x32_f16(false, a1, false, b0, (short)0, c10, false, false);
        c11 = __builtin_amdgcn_wmma_f32_16x16x32_f16(false, a1, false, b1, (short)0, c11, false, false);
        buf ^= 1;
    }

    // epilogue: D layout -> col n = lane%16, row = vgpr + (lane>=16)*8
    const int n0 = ntile + wn * 32 + lrow;
    const int n1 = n0 + 16;
    const float bi0 = bias ? bias[n0] : 0.0f;
    const float bi1 = bias ? bias[n1] : 0.0f;
    #pragma unroll
    for (int rr = 0; rr < 8; ++rr) {
        int m0 = mtile + wm * 32 + lhi * 8 + rr;
        int m1 = m0 + 16;
        float v00 = c00[rr] + bi0, v01 = c01[rr] + bi1;
        float v10 = c10[rr] + bi0, v11 = c11[rr] + bi1;
        if (ACT == 1) {
            v00 = gelu_exact(v00); v01 = gelu_exact(v01);
            v10 = gelu_exact(v10); v11 = gelu_exact(v11);
        }
        if (OUT_F16) {
            _Float16* Co = (_Float16*)Cptr;
            if (!GUARD || m0 < M) { Co[(size_t)m0*N + n0] = (_Float16)v00; Co[(size_t)m0*N + n1] = (_Float16)v01; }
            if (!GUARD || m1 < M) { Co[(size_t)m1*N + n0] = (_Float16)v10; Co[(size_t)m1*N + n1] = (_Float16)v11; }
        } else {
            float* Co = (float*)Cptr;
            if (!GUARD || m0 < M) { Co[(size_t)m0*N + n0] = v00; Co[(size_t)m0*N + n1] = v01; }
            if (!GUARD || m1 < M) { Co[(size_t)m1*N + n0] = v10; Co[(size_t)m1*N + n1] = v11; }
        }
    }
}

// =====================================================================
// fused residual + LayerNorm: x = LN(x + y)*g + b; also emits f16 shadow
// =====================================================================
__global__ __launch_bounds__(256)
void add_ln_kernel(float* __restrict__ x, _Float16* __restrict__ xh,
                   const float* __restrict__ y,
                   const float* __restrict__ g, const float* __restrict__ b)
{
    __shared__ float sh[256];
    const int tok = blockIdx.x;
    const int c   = threadIdx.x;
    const size_t base = (size_t)tok * CDIM;
    float v = x[base + c] + y[base + c];

    sh[c] = v; __syncthreads();
    #pragma unroll
    for (int s = 128; s > 0; s >>= 1) { if (c < s) sh[c] += sh[c + s]; __syncthreads(); }
    float mean = sh[0] * (1.0f / CDIM);
    __syncthreads();
    float d = v - mean;
    sh[c] = d * d; __syncthreads();
    #pragma unroll
    for (int s = 128; s > 0; s >>= 1) { if (c < s) sh[c] += sh[c + s]; __syncthreads(); }
    float var = sh[0] * (1.0f / CDIM);
    float o = d * rsqrtf(var + 1e-5f) * g[c] + b[c];
    x[base + c]  = o;
    xh[base + c] = (_Float16)o;
}

// =====================================================================
// cross-attention: block per pixel, wave32 per head, lane = head dim.
// OOB neighbours contribute only the positional projection (zero-pad
// before pos-add in the reference). Output f16 (feeds out-proj GEMM).
// =====================================================================
__global__ __launch_bounds__(256)
void attn_kernel(const float* __restrict__ q,
                 const _Float16* __restrict__ Kf, const _Float16* __restrict__ Vf,
                 const float* __restrict__ posK, const float* __restrict__ posV,
                 _Float16* __restrict__ out)
{
    const int pix  = blockIdx.x;
    const int tid  = threadIdx.x;
    const int head = tid >> 5;
    const int lane = tid & 31;
    const int b = pix >> 10;
    const int y = (pix >> 5) & (HH - 1);
    const int x = pix & (WW - 1);
    const int c = head * HD + lane;

    const float qv = q[(size_t)pix * CDIM + c] * 0.17677669529663687f; // 1/sqrt(32)

    float sc[LK];
    #pragma unroll
    for (int tk = 0; tk < LK; ++tk) {
        int t = tk / K2, k = tk % K2;
        int y2 = y + (k / KW) - 1;
        int x2 = x + (k % KW) - 1;
        float kval = posK[tk * CDIM + c];
        if (y2 >= 0 && y2 < HH && x2 >= 0 && x2 < WW) {
            long m = ((long)(b * TT + t) * HH + y2) * WW + x2;
            kval += (float)Kf[m * CDIM + c];
        }
        float s = qv * kval;
        #pragma unroll
        for (int off = 16; off > 0; off >>= 1) s += __shfl_xor(s, off, 32);
        sc[tk] = s;
    }
    float mx = -1e30f;
    #pragma unroll
    for (int tk = 0; tk < LK; ++tk) mx = fmaxf(mx, sc[tk]);
    float den = 0.0f;
    #pragma unroll
    for (int tk = 0; tk < LK; ++tk) { sc[tk] = __expf(sc[tk] - mx); den += sc[tk]; }
    const float inv = 1.0f / den;

    float acc = 0.0f;
    #pragma unroll
    for (int tk = 0; tk < LK; ++tk) {
        int t = tk / K2, k = tk % K2;
        int y2 = y + (k / KW) - 1;
        int x2 = x + (k % KW) - 1;
        float vval = posV[tk * CDIM + c];
        if (y2 >= 0 && y2 < HH && x2 >= 0 && x2 < WW) {
            long m = ((long)(b * TT + t) * HH + y2) * WW + x2;
            vval += (float)Vf[m * CDIM + c];
        }
        acc += sc[tk] * inv * vval;
    }
    out[(size_t)pix * CDIM + c] = (_Float16)acc;
}

// final: x (B*H*W, C) -> out (B, C, H, W)
__global__ __launch_bounds__(256)
void out_kernel(const float* __restrict__ x, float* __restrict__ out)
{
    long i = (long)blockIdx.x * blockDim.x + threadIdx.x;
    if (i >= (long)BB*CDIM*HH*WW) return;
    int w = (int)(i & (WW-1));
    int h = (int)((i >> 5) & (HH-1));
    int c = (int)((i >> 10) & (CDIM-1));
    int b = (int)(i >> 18);
    out[i] = x[((size_t)(b*(HH*WW) + h*WW + w))*CDIM + c];
}

// ---------------------------------------------------------------------
static void run_gemm(hipStream_t s, const _Float16* A, const _Float16* Wh,
                     const float* bias, void* C, int out_f16, int act, int guard,
                     int M, int N, int K)
{
    dim3 grid(N / 64, (M + 127) / 128);
    dim3 blk(256);
    if (guard)
        gemm_wmma<false, 0, true ><<<grid, blk, 0, s>>>(A, Wh, bias, C, M, N, K);
    else if (out_f16 && act == 1)
        gemm_wmma<true,  1, false><<<grid, blk, 0, s>>>(A, Wh, bias, C, M, N, K);
    else if (out_f16)
        gemm_wmma<true,  0, false><<<grid, blk, 0, s>>>(A, Wh, bias, C, M, N, K);
    else
        gemm_wmma<false, 0, false><<<grid, blk, 0, s>>>(A, Wh, bias, C, M, N, K);
}

static void cvt(hipStream_t s, const float* src, _Float16* dst, int n)
{
    cvt_f16_kernel<<<(n + 255) / 256, 256, 0, s>>>(src, dst, n);
}

extern "C" void kernel_launch(void* const* d_in, const int* in_sizes, int n_in,
                              void* d_out, int out_size, void* d_ws, size_t ws_size,
                              hipStream_t stream)
{
    (void)in_sizes; (void)n_in; (void)out_size; (void)ws_size;

    const float* feats = (const float*)d_in[0];
    const float* tp    = (const float*)d_in[1];
    const float* sp    = (const float*)d_in[2];

    // ---------------- workspace layout (bytes) ----------------
    char* ws = (char*)d_ws;
    _Float16* featsT  = (_Float16*)(ws);                 // 20480*256 f16
    _Float16* Kf      = (_Float16*)(ws + 10485760);
    _Float16* Vf      = (_Float16*)(ws + 20971520);
    float*    x       = (float*)   (ws + 31457280);      // 4096*256 f32
    _Float16* xh      = (_Float16*)(ws + 35651584);      // f16 shadow of x
    float*    q       = (float*)   (ws + 37748736);
    _Float16* hiddenh = (_Float16*)(ws + 41943040);      // 4096*1024 f16 (FFN)
    _Float16* attnh   = (_Float16*)(ws + 50331648);      // 4096*256 f16
    _Float16* vsah    = (_Float16*)(ws + 52428800);      // 4096*256 f16 (SA value)
    float*    buff32  = (float*)   (ws + 54525952);      // 4096*256 f32 (residual in)
    _Float16* posmh   = (_Float16*)(ws + 58720256);      // 45*256 f16
    float*    posK    = (float*)   (ws + 58743296);
    float*    posV    = (float*)   (ws + 58789376);
    _Float16* wh      = (_Float16*)(ws + 58835456);      // f16 weights, 2 layers

    // per-layer f16 weight sub-offsets (elements)
    const int W_SAV = 0,        W_SAO = 65536,  W_CAI = 131072;
    const int W_CAO = 327680,   W_L1  = 393216, W_L2  = 655360;
    const int W_LAYER = 917504;

    // ---- preprocessing ----
    {
        long n = (long)BB*TT*CDIM*HH*WW;
        prep_kernel<<<(unsigned)((n + 255) / 256), 256, 0, stream>>>(feats, featsT, x, xh);
        posmat_kernel<<<(LK*CDIM + 255) / 256, 256, 0, stream>>>(tp, sp, posmh);
    }

    for (int l = 0; l < 2; ++l) {
        const int pb = 3 + 18 * l;
        const float* sa_w_in  = (const float*)d_in[pb + 0];
        const float* sa_b_in  = (const float*)d_in[pb + 1];
        const float* sa_w_out = (const float*)d_in[pb + 2];
        const float* sa_b_out = (const float*)d_in[pb + 3];
        const float* ca_w_in  = (const float*)d_in[pb + 4];
        const float* ca_b_in  = (const float*)d_in[pb + 5];
        const float* ca_w_out = (const float*)d_in[pb + 6];
        const float* ca_b_out = (const float*)d_in[pb + 7];
        const float* lin1_w   = (const float*)d_in[pb + 8];
        const float* lin1_b   = (const float*)d_in[pb + 9];
        const float* lin2_w   = (const float*)d_in[pb + 10];
        const float* lin2_b   = (const float*)d_in[pb + 11];
        const float* ln1_g    = (const float*)d_in[pb + 12];
        const float* ln1_b    = (const float*)d_in[pb + 13];
        const float* ln2_g    = (const float*)d_in[pb + 14];
        const float* ln2_b    = (const float*)d_in[pb + 15];
        const float* ln3_g    = (const float*)d_in[pb + 16];
        const float* ln3_b    = (const float*)d_in[pb + 17];

        _Float16* whl = wh + (size_t)l * W_LAYER;
        cvt(stream, sa_w_in + 2*CDIM*CDIM, whl + W_SAV, CDIM*CDIM);   // only Wv of SA is live
        cvt(stream, sa_w_out,              whl + W_SAO, CDIM*CDIM);
        cvt(stream, ca_w_in,               whl + W_CAI, 3*CDIM*CDIM);
        cvt(stream, ca_w_out,              whl + W_CAO, CDIM*CDIM);
        cvt(stream, lin1_w,                whl + W_L1,  4*CDIM*CDIM);
        cvt(stream, lin2_w,                whl + W_L2,  4*CDIM*CDIM);
        const _Float16* ca_q_h = whl + W_CAI;
        const _Float16* ca_k_h = whl + W_CAI + CDIM*CDIM;
        const _Float16* ca_v_h = whl + W_CAI + 2*CDIM*CDIM;

        // ---- self-attention over a length-1 sequence (softmax(1x1)=1):
        //      sa = (x @ Wv^T + bv) @ Wout^T + bout   (exact) ----
        run_gemm(stream, xh,   whl + W_SAV, sa_b_in + 2*CDIM, vsah,  1, 0, 0, NPIX, CDIM, CDIM);
        run_gemm(stream, vsah, whl + W_SAO, sa_b_out,         buff32,0, 0, 0, NPIX, CDIM, CDIM);
        add_ln_kernel<<<NPIX, CDIM, 0, stream>>>(x, xh, buff32, ln1_g, ln1_b);

        // ---- cross-attention ----
        run_gemm(stream, xh,     ca_q_h, ca_b_in,          q,    0, 0, 0, NPIX,  CDIM, CDIM);
        run_gemm(stream, featsT, ca_k_h, nullptr,          Kf,   1, 0, 0, MFEAT, CDIM, CDIM);
        run_gemm(stream, featsT, ca_v_h, nullptr,          Vf,   1, 0, 0, MFEAT, CDIM, CDIM);
        run_gemm(stream, posmh,  ca_k_h, ca_b_in +   CDIM, posK, 0, 0, 1, LK,    CDIM, CDIM);
        run_gemm(stream, posmh,  ca_v_h, ca_b_in + 2*CDIM, posV, 0, 0, 1, LK,    CDIM, CDIM);
        attn_kernel<<<NPIX, 256, 0, stream>>>(q, Kf, Vf, posK, posV, attnh);
        run_gemm(stream, attnh, whl + W_CAO, ca_b_out, buff32, 0, 0, 0, NPIX, CDIM, CDIM);
        add_ln_kernel<<<NPIX, CDIM, 0, stream>>>(x, xh, buff32, ln2_g, ln2_b);

        // ---- FFN: GELU fused into first GEMM, f16 hidden ----
        run_gemm(stream, xh,      whl + W_L1, lin1_b, hiddenh, 1, 1, 0, NPIX, 4*CDIM, CDIM);
        run_gemm(stream, hiddenh, whl + W_L2, lin2_b, buff32,  0, 0, 0, NPIX, CDIM, 4*CDIM);
        add_ln_kernel<<<NPIX, CDIM, 0, stream>>>(x, xh, buff32, ln3_g, ln3_b);
    }

    out_kernel<<<(BB*CDIM*HH*WW + 255) / 256, 256, 0, stream>>>(x, (float*)d_out);
}
